// CrossAttention_1726576855890
// MI455X (gfx1250) — matrix operand order repaired and checked by reading, not compiled, not run on hardware
//
#include <hip/hip_runtime.h>

// Criss-cross attention block for MI455X (gfx1250), bf16 WMMA + fp32 accumulate.
// All WMMA fragments are built from pair-contiguous memory: each of the 8
// fragment VGPRs fills with a single aligned b32 load (no half-register
// packing movs). X and the attention output are stored channel-pair
// interleaved ([q][p][2]); LDS tiles for scores/apply are staged the same way.

typedef __attribute__((ext_vector_type(16))) __bf16   v16bf;
typedef __attribute__((ext_vector_type(8)))  float    v8f;
typedef __attribute__((ext_vector_type(8)))  unsigned v8u;

union Frag { v16bf bf; v8u u; };

#define N_B      4
#define C_IN_    512
#define C_INNER_ 64
#define C_OUT_   512
#define H_DIM    128
#define W_DIM    128
#define P_PIX    (H_DIM * W_DIM)   // 16384
#define PAD_     136               // padded LDS row (bf16) for G tiles
#define ARS_     258               // prob-tile pair-row stride (129 dwords)

__device__ __forceinline__ unsigned short bfbits(float f) {
  unsigned u = __builtin_bit_cast(unsigned, f);
  return (unsigned short)((u + 0x7FFFu + ((u >> 16) & 1u)) >> 16);
}
__device__ __forceinline__ __bf16 f2bf(float f) {
  return __builtin_bit_cast(__bf16, bfbits(f));
}

#define WMMA_BF16(c, a, b) \
  (c) = __builtin_amdgcn_wmma_f32_16x16x32_bf16(false, (a), false, (b), (short)0, (c), false, false)

// ---- fragment loaders (each VGPR = one b32 load; base holds runtime terms) --
// A row-major, k contiguous: srcLane = src + m*ld + half*8   (+ kk outside)
__device__ __forceinline__ v16bf a_frag_pairs(const __bf16* srcLane) {
  Frag a;
#pragma unroll
  for (int j = 0; j < 8; ++j)
    a.u[j] = *(const unsigned*)(srcLane + ((j & 3) * 2 + (j >> 2) * 16));
  return a.bf;
}
// A from pair-interleaved col-major tile (pair-row stride RS elems):
// srcLane = tile + m*2 + half*4*RS   (+ (kk/2)*RS outside)
template <int RS>
__device__ __forceinline__ v16bf a_frag_pairs_colmaj(const __bf16* srcLane) {
  Frag a;
#pragma unroll
  for (int j = 0; j < 8; ++j)
    a.u[j] = *(const unsigned*)(srcLane + (size_t)((j & 3) + (j >> 2) * 8) * RS);
  return a.bf;
}
// B from pair-interleaved tile (pair-row stride RS elems):
// srcLane = tile + n*2 + half*4*RS   (+ (kk/2)*RS outside)
template <int RS>
__device__ __forceinline__ v16bf b_frag_pairs(const __bf16* srcLane) {
  Frag b;
#pragma unroll
  for (int j = 0; j < 8; ++j)
    b.u[j] = *(const unsigned*)(srcLane + (size_t)((j & 3) + (j >> 2) * 8) * RS);
  return b.bf;
}
// C store: dstLane = dst + (m0 + half*8)*LDN + n0 + nl
template <int LDN>
__device__ __forceinline__ void c_store(float* dstLane, v8f c) {
#pragma unroll
  for (int r = 0; r < 8; ++r) dstLane[(size_t)r * LDN] = c[r];
}

// ---------------- Kernel 0a: plain f32 -> bf16 (weights) ---------------------
__global__ __launch_bounds__(256) void k_f32_to_bf16(
    const float* __restrict__ src, __bf16* __restrict__ dst) {
  const size_t i0 = ((size_t)blockIdx.x * 256 + threadIdx.x) * 8;
  float4 a = *(const float4*)(src + i0);
  float4 b = *(const float4*)(src + i0 + 4);
  uint4 o;
  o.x = (unsigned)bfbits(a.x) | ((unsigned)bfbits(a.y) << 16);
  o.y = (unsigned)bfbits(a.z) | ((unsigned)bfbits(a.w) << 16);
  o.z = (unsigned)bfbits(b.x) | ((unsigned)bfbits(b.y) << 16);
  o.w = (unsigned)bfbits(b.z) | ((unsigned)bfbits(b.w) << 16);
  *(uint4*)(dst + i0) = o;
}

// ---------------- Kernel 0b: f32 [2q][p],[2q+1][p] -> bf16 [q][p][2] ---------
__global__ __launch_bounds__(256) void k_cvt_interleave(
    const float* __restrict__ src, __bf16* __restrict__ dst) {
  const size_t t = (size_t)blockIdx.x * 256 + threadIdx.x;
  const size_t q = t >> 12;                 // / (P_PIX/4)
  const int    po = (int)(t & 4095) * 4;    // pixel offset
  const float* r0 = src + q * (2 * P_PIX) + po;
  float4 a = *(const float4*)r0;
  float4 b = *(const float4*)(r0 + P_PIX);
  uint4 o;
  o.x = (unsigned)bfbits(a.x) | ((unsigned)bfbits(b.x) << 16);
  o.y = (unsigned)bfbits(a.y) | ((unsigned)bfbits(b.y) << 16);
  o.z = (unsigned)bfbits(a.z) | ((unsigned)bfbits(b.z) << 16);
  o.w = (unsigned)bfbits(a.w) | ((unsigned)bfbits(b.w) << 16);
  *(uint4*)(dst + q * (2 * P_PIX) + 2 * po) = o;
}

// ---------------- Kernel 1: fused t/f/g 1x1 conv (GEMM 640 x 512 x P) -------
__global__ __launch_bounds__(256) void k_conv_tfg(
    const __bf16* __restrict__ Xp,
    const __bf16* __restrict__ Wt, const __bf16* __restrict__ Wf, const __bf16* __restrict__ Wg,
    const float* __restrict__ tb, const float* __restrict__ fb, const float* __restrict__ gb,
    __bf16* __restrict__ T, __bf16* __restrict__ F, __bf16* __restrict__ G) {
  const int lane = threadIdx.x & 31, wv = threadIdx.x >> 5;
  const int half = (lane >> 4) & 1, m = lane & 15, nl = lane & 15;
  const int ptile = blockIdx.x * 8 + wv;   // 0..1023
  const int mt = blockIdx.y;               // 0..39 (t:0-3, f:4-7, g:8-39)
  const int n = blockIdx.z;
  const int p0 = ptile * 16;

  const __bf16* wsrc; const float* bsrc; __bf16* dst; int orow;
  if (mt < 4)      { wsrc = Wt; bsrc = tb; dst = T + (size_t)n * C_INNER_ * P_PIX; orow = mt * 16; }
  else if (mt < 8) { wsrc = Wf; bsrc = fb; dst = F + (size_t)n * C_INNER_ * P_PIX; orow = (mt - 4) * 16; }
  else             { wsrc = Wg; bsrc = gb; dst = G + (size_t)n * C_OUT_  * P_PIX; orow = (mt - 8) * 16; }

  const __bf16* aLane = wsrc + (size_t)(orow + m) * C_IN_ + half * 8;
  const __bf16* bLane = Xp + (size_t)n * C_IN_ * P_PIX
                      + (size_t)(half * 8) * P_PIX + 2 * (p0 + nl);

  v8f acc = {};
  for (int kk = 0; kk < C_IN_; kk += 32) {
    v16bf a = a_frag_pairs(aLane + kk);
    v16bf b = b_frag_pairs<2 * P_PIX>(bLane + (size_t)kk * P_PIX);
    WMMA_BF16(acc, a, b);
  }
  __bf16* dLane = dst + (size_t)(orow + half * 8) * P_PIX + p0 + nl;
  const float* biLane = bsrc + orow + half * 8;
#pragma unroll
  for (int r = 0; r < 8; ++r)
    dLane[(size_t)r * P_PIX] = f2bf(acc[r] + biLane[r]);
}

// ---------------- Kernel 2: row scores e_row[n,h,w,v] ------------------------
__global__ __launch_bounds__(256) void k_row_scores(
    const __bf16* __restrict__ T, const __bf16* __restrict__ F, float* __restrict__ Erow) {
  __shared__ __align__(16) __bf16 Tl[32 * 256];   // pair-interleaved [q][w][2]
  __shared__ __align__(16) __bf16 Fl[32 * 256];
  const int h = blockIdx.x, n = blockIdx.y;
  const int tid = threadIdx.x, lane = tid & 31, wv = tid >> 5;
  const int half = (lane >> 4) & 1, m = lane & 15, nl = lane & 15;
  const __bf16* Tg = T + (size_t)n * C_INNER_ * P_PIX + (size_t)h * W_DIM;
  const __bf16* Fg = F + (size_t)n * C_INNER_ * P_PIX + (size_t)h * W_DIM;
#pragma unroll
  for (int j = 0; j < 4; ++j) {           // 1024 chunks: 4 pixels x 2 channels
    int ch = tid + j * 256; int q = ch >> 5, po = (ch & 31) * 4;
    uint2 a = *(const uint2*)(Tg + (size_t)(2 * q) * P_PIX + po);
    uint2 b = *(const uint2*)(Tg + (size_t)(2 * q + 1) * P_PIX + po);
    uint4 o;
    o.x = (a.x & 0xFFFFu) | (b.x << 16);
    o.y = (a.x >> 16) | (b.x & 0xFFFF0000u);
    o.z = (a.y & 0xFFFFu) | (b.y << 16);
    o.w = (a.y >> 16) | (b.y & 0xFFFF0000u);
    *(uint4*)&Tl[q * 256 + po * 2] = o;
    a = *(const uint2*)(Fg + (size_t)(2 * q) * P_PIX + po);
    b = *(const uint2*)(Fg + (size_t)(2 * q + 1) * P_PIX + po);
    o.x = (a.x & 0xFFFFu) | (b.x << 16);
    o.y = (a.x >> 16) | (b.x & 0xFFFF0000u);
    o.z = (a.y & 0xFFFFu) | (b.y << 16);
    o.w = (a.y >> 16) | (b.y & 0xFFFF0000u);
    *(uint4*)&Fl[q * 256 + po * 2] = o;
  }
  __syncthreads();
  float* Eb = Erow + ((size_t)n * P_PIX + (size_t)h * W_DIM) * W_DIM;  // [w][v]
  const int mt = wv;
  const __bf16* aL0 = Tl + (mt * 16 + m) * 2 + half * 4 * 256;   // A[w][c]=T[c][w]
  for (int nt = 0; nt < 8; ++nt) {
    const __bf16* bL0 = Fl + (nt * 16 + nl) * 2 + half * 4 * 256;
    v8f acc = {};
#pragma unroll
    for (int kk = 0; kk < C_INNER_; kk += 32) {
      v16bf a = a_frag_pairs_colmaj<256>(aL0 + kk * 128);
      v16bf b = b_frag_pairs<256>(bL0 + kk * 128);
      WMMA_BF16(acc, a, b);
    }
    c_store<W_DIM>(Eb + (size_t)(mt * 16 + half * 8) * W_DIM + nt * 16 + nl, acc);
  }
}

// ---------------- Kernel 3: col scores e_col[n,h,w,u] ------------------------
__global__ __launch_bounds__(256) void k_col_scores(
    const __bf16* __restrict__ T, const __bf16* __restrict__ F, float* __restrict__ Ecol) {
  __shared__ __align__(16) __bf16 Tl[32 * 256];   // pair-interleaved [q][h][2]
  __shared__ __align__(16) __bf16 Fl[32 * 256];
  const int w = blockIdx.x, n = blockIdx.y;
  const int tid = threadIdx.x, lane = tid & 31, wv = tid >> 5;
  const int half = (lane >> 4) & 1, m = lane & 15, nl = lane & 15;
  const __bf16* Tg = T + (size_t)n * C_INNER_ * P_PIX + (size_t)w;
  const __bf16* Fg = F + (size_t)n * C_INNER_ * P_PIX + (size_t)w;
  for (int j = 0; j < 16; ++j) {          // 4096 u32 writes (strided gather)
    int ch = tid + j * 256; int hh = ch & 127, q = ch >> 7;
    unsigned short t0 = __builtin_bit_cast(unsigned short,
        Tg[(size_t)(2 * q) * P_PIX + (size_t)hh * W_DIM]);
    unsigned short t1 = __builtin_bit_cast(unsigned short,
        Tg[(size_t)(2 * q + 1) * P_PIX + (size_t)hh * W_DIM]);
    *(unsigned*)&Tl[q * 256 + hh * 2] = (unsigned)t0 | ((unsigned)t1 << 16);
    unsigned short f0 = __builtin_bit_cast(unsigned short,
        Fg[(size_t)(2 * q) * P_PIX + (size_t)hh * W_DIM]);
    unsigned short f1 = __builtin_bit_cast(unsigned short,
        Fg[(size_t)(2 * q + 1) * P_PIX + (size_t)hh * W_DIM]);
    *(unsigned*)&Fl[q * 256 + hh * 2] = (unsigned)f0 | ((unsigned)f1 << 16);
  }
  __syncthreads();
  float* Eb = Ecol + ((size_t)n * P_PIX + (size_t)w) * W_DIM;  // (h,u) at h*16384 + u
  const int mt = wv;
  const __bf16* aL0 = Tl + (mt * 16 + m) * 2 + half * 4 * 256;  // A[h][c]=T[c][h]
  for (int nt = 0; nt < 8; ++nt) {
    const __bf16* bL0 = Fl + (nt * 16 + nl) * 2 + half * 4 * 256;
    v8f acc = {};
#pragma unroll
    for (int kk = 0; kk < C_INNER_; kk += 32) {
      v16bf a = a_frag_pairs_colmaj<256>(aL0 + kk * 128);
      v16bf b = b_frag_pairs<256>(bL0 + kk * 128);
      WMMA_BF16(acc, a, b);
    }
    c_store<W_DIM * 128>(Eb + (size_t)(mt * 16 + half * 8) * (W_DIM * 128) + nt * 16 + nl, acc);
  }
}

// ---------------- Kernel 4: softmax over 256 logits per pixel (wave/pixel) ---
__global__ __launch_bounds__(256) void k_softmax(float* __restrict__ Erow, float* __restrict__ Ecol) {
  const int lane = threadIdx.x & 31, wv = threadIdx.x >> 5;
  const size_t pix = (size_t)blockIdx.x * 8 + wv;
  const int h = (int)((pix >> 7) & 127);
  float* er = Erow + pix * 128;
  float* ec = Ecol + pix * 128;
  float v[8];
#pragma unroll
  for (int j = 0; j < 8; ++j) {
    int idx = lane * 8 + j;
    if (idx < 128) v[j] = er[idx];
    else { int u = idx - 128; v[j] = (u == h) ? -3.0e38f : ec[u]; }
  }
  float mx = v[0];
#pragma unroll
  for (int j = 1; j < 8; ++j) mx = fmaxf(mx, v[j]);
#pragma unroll
  for (int o = 16; o > 0; o >>= 1) mx = fmaxf(mx, __shfl_xor(mx, o, 32));
  float s = 0.f;
#pragma unroll
  for (int j = 0; j < 8; ++j) { v[j] = __expf(v[j] - mx); s += v[j]; }
#pragma unroll
  for (int o = 16; o > 0; o >>= 1) s += __shfl_xor(s, o, 32);
  const float inv = 1.0f / s;
#pragma unroll
  for (int j = 0; j < 8; ++j) {
    int idx = lane * 8 + j; float p = v[j] * inv;
    if (idx < 128) er[idx] = p;
    else { int u = idx - 128; ec[u] = (u == h) ? 0.0f : p; }
  }
}

// ---------------- Kernel 5: row apply: Out[c,w] = sum_v Arow[w,v]*G[c,h,v] ---
// One block per (n,h): probs staged+converted ONCE, then loop over 8 c-chunks.
__global__ __launch_bounds__(256) void k_apply_row(
    const __bf16* __restrict__ G, const float* __restrict__ Arow, float* __restrict__ Out) {
  __shared__ __align__(16) __bf16 Gl[64 * PAD_];   // [c][v] row-major
  __shared__ __align__(16) __bf16 Al[64 * ARS_];   // pair-interleaved [q=v/2][w][2]
  const int h = blockIdx.x, n = blockIdx.y;
  const int tid = threadIdx.x, lane = tid & 31, wv = tid >> 5;
  const int half = (lane >> 4) & 1, m = lane & 15, nl = lane & 15;
  const float* Ab = Arow + ((size_t)n * P_PIX + (size_t)h * W_DIM) * 128;  // [w][v]
  for (int j = 0; j < 16; ++j) {          // 4096 chunks of 4 floats (transpose)
    int ch = tid + j * 256; int w = ch >> 5, vo = (ch & 31) * 4;
    const float* p = Ab + (size_t)w * 128 + vo;
    *(unsigned*)&Al[(vo >> 1) * ARS_ + w * 2] =
        (unsigned)bfbits(p[0]) | ((unsigned)bfbits(p[1]) << 16);
    *(unsigned*)&Al[((vo >> 1) + 1) * ARS_ + w * 2] =
        (unsigned)bfbits(p[2]) | ((unsigned)bfbits(p[3]) << 16);
  }
  const __bf16* Gg0 = G + (size_t)n * C_OUT_ * P_PIX + (size_t)h * W_DIM;
  float* Ob0 = Out + (size_t)n * C_OUT_ * P_PIX + (size_t)h * W_DIM;
  const int mt = wv & 3, ntb = (wv >> 2) * 4;
  for (int cc = 0; cc < 8; ++cc) {
    __syncthreads();                      // Gl free (also covers Al staging, cc==0)
    const __bf16* Gg = Gg0 + (size_t)cc * 64 * P_PIX;
#pragma unroll
    for (int j = 0; j < 4; ++j) {         // 1024 chunks of 8 bf16
      int ch = tid + j * 256; int ci = ch >> 4, vo = (ch & 15) * 8;
      *(uint4*)&Gl[ci * PAD_ + vo] = *(const uint4*)(Gg + (size_t)ci * P_PIX + vo);
    }
    __syncthreads();
    const __bf16* aL0 = Gl + (size_t)(mt * 16 + m) * PAD_ + half * 8;
    float* Ob = Ob0 + (size_t)cc * 64 * P_PIX;
    for (int i = 0; i < 4; ++i) {
      int nt = ntb + i;
      const __bf16* bL0 = Al + (nt * 16 + nl) * 2 + half * 4 * ARS_;
      v8f acc = {};
#pragma unroll
      for (int kk = 0; kk < 128; kk += 32) {
        v16bf a = a_frag_pairs(aL0 + kk);
        v16bf b = b_frag_pairs<ARS_>(bL0 + kk * (ARS_ / 2));
        WMMA_BF16(acc, a, b);
      }
      c_store<P_PIX>(Ob + (size_t)(mt * 16 + half * 8) * P_PIX + nt * 16 + nl, acc);
    }
  }
}

// ---------------- Kernel 6: col apply: Out[c,h] += sum_u Acol[h,u]*G[c,u,w] --
__global__ __launch_bounds__(256) void k_apply_col(
    const __bf16* __restrict__ G, const float* __restrict__ Acol, float* __restrict__ Out) {
  __shared__ __align__(16) __bf16 Gl[64 * PAD_];   // [c][u] row-major
  __shared__ __align__(16) __bf16 Al[64 * ARS_];   // pair-interleaved [q=u/2][h][2]
  const int w = blockIdx.x, n = blockIdx.y;
  const int tid = threadIdx.x, lane = tid & 31, wv = tid >> 5;
  const int half = (lane >> 4) & 1, m = lane & 15, nl = lane & 15;
  const float* Ab = Acol + ((size_t)n * P_PIX + (size_t)w) * 128;  // (h,u) at h*16384 + u
  for (int j = 0; j < 16; ++j) {          // 4096 chunks of 4 floats (transpose)
    int ch = tid + j * 256; int hh = ch >> 5, uo = (ch & 31) * 4;
    const float* p = Ab + (size_t)hh * (W_DIM * 128) + uo;
    *(unsigned*)&Al[(uo >> 1) * ARS_ + hh * 2] =
        (unsigned)bfbits(p[0]) | ((unsigned)bfbits(p[1]) << 16);
    *(unsigned*)&Al[((uo >> 1) + 1) * ARS_ + hh * 2] =
        (unsigned)bfbits(p[2]) | ((unsigned)bfbits(p[3]) << 16);
  }
  const __bf16* Gg0 = G + (size_t)n * C_OUT_ * P_PIX + (size_t)w;
  float* Ob0 = Out + (size_t)n * C_OUT_ * P_PIX + (size_t)w;
  const int mt = wv & 3, ntb = (wv >> 2) * 4;
  for (int cc = 0; cc < 8; ++cc) {
    __syncthreads();
    const __bf16* Gg = Gg0 + (size_t)cc * 64 * P_PIX;
    for (int j = 0; j < (64 * 128) / 256; ++j) {   // strided gather (stride W)
      int idx = tid + j * 256; int ci = idx >> 7, u = idx & 127;
      Gl[ci * PAD_ + u] = Gg[(size_t)ci * P_PIX + (size_t)u * W_DIM];
    }
    __syncthreads();
    const __bf16* aL0 = Gl + (size_t)(mt * 16 + m) * PAD_ + half * 8;
    float* Ob = Ob0 + (size_t)cc * 64 * P_PIX;
    for (int i = 0; i < 4; ++i) {
      int nt = ntb + i;
      const __bf16* bL0 = Al + (nt * 16 + nl) * 2 + half * 4 * ARS_;
      v8f acc = {};
#pragma unroll
      for (int kk = 0; kk < 128; kk += 32) {
        v16bf a = a_frag_pairs(aL0 + kk);
        v16bf b = b_frag_pairs<ARS_>(bL0 + kk * (ARS_ / 2));
        WMMA_BF16(acc, a, b);
      }
      float* oL = Ob + (size_t)(mt * 16 + half * 8) * P_PIX + (size_t)(nt * 16 + nl) * W_DIM;
#pragma unroll
      for (int r = 0; r < 8; ++r)
        oL[(size_t)r * P_PIX] += acc[r];   // disjoint across blocks; after k_apply_row
    }
  }
}

// ---------------- Kernel 7: final 1x1 conv + residual ------------------------
__global__ __launch_bounds__(256) void k_final(
    const float* __restrict__ x, const __bf16* __restrict__ Winc, const float* __restrict__ ib,
    const __bf16* __restrict__ Op, float* __restrict__ y) {
  const int lane = threadIdx.x & 31, wv = threadIdx.x >> 5;
  const int half = (lane >> 4) & 1, m = lane & 15, nl = lane & 15;
  const int ptile = blockIdx.x * 8 + wv, mt = blockIdx.y, n = blockIdx.z;
  const int p0 = ptile * 16;
  const __bf16* aLane = Winc + (size_t)(mt * 16 + m) * C_OUT_ + half * 8;
  const __bf16* bLane = Op + (size_t)n * C_OUT_ * P_PIX
                      + (size_t)(half * 8) * P_PIX + 2 * (p0 + nl);
  v8f acc = {};
  for (int kk = 0; kk < C_OUT_; kk += 32) {
    v16bf a = a_frag_pairs(aLane + kk);
    v16bf b = b_frag_pairs<2 * P_PIX>(bLane + (size_t)kk * P_PIX);
    WMMA_BF16(acc, a, b);
  }
  const float* xL = x + ((size_t)n * C_IN_ + mt * 16 + half * 8) * P_PIX + p0 + nl;
  float* yL = y + ((size_t)n * C_IN_ + mt * 16 + half * 8) * P_PIX + p0 + nl;
  const float* biL = ib + mt * 16 + half * 8;
#pragma unroll
  for (int r = 0; r < 8; ++r)
    yL[(size_t)r * P_PIX] = xL[(size_t)r * P_PIX] + acc[r] + biL[r];
}

// ---------------- Launcher ---------------------------------------------------
extern "C" void kernel_launch(void* const* d_in, const int* in_sizes, int n_in,
                              void* d_out, int out_size, void* d_ws, size_t ws_size,
                              hipStream_t stream) {
  const float* x    = (const float*)d_in[0];
  const float* t_w  = (const float*)d_in[1];
  const float* t_b  = (const float*)d_in[2];
  const float* f_w  = (const float*)d_in[3];
  const float* f_b  = (const float*)d_in[4];
  const float* g_w  = (const float*)d_in[5];
  const float* g_b  = (const float*)d_in[6];
  const float* iw   = (const float*)d_in[7];
  const float* ibv  = (const float*)d_in[8];
  float* y = (float*)d_out;

  const size_t NP = (size_t)N_B * P_PIX;
  __bf16* T    = (__bf16*)d_ws;
  __bf16* F    = T + NP * C_INNER_;
  __bf16* G    = F + NP * C_INNER_;
  __bf16* Xp   = G + NP * C_OUT_;           // pair-interleaved x
  __bf16* Op   = Xp + NP * C_IN_;           // pair-interleaved attention out
  __bf16* Wt   = Op + NP * C_OUT_;
  __bf16* Wf   = Wt + (size_t)C_INNER_ * C_IN_;
  __bf16* Wg   = Wf + (size_t)C_INNER_ * C_IN_;
  __bf16* Winc = Wg + (size_t)C_OUT_ * C_IN_;
  float* Erow  = (float*)(Winc + (size_t)C_IN_ * C_OUT_);
  float* Ecol  = Erow + NP * 128;
  float* Out   = Ecol + NP * 128;
  // total ws ≈ 219 MB bf16 + 201 MB fp32 ≈ 420 MB

  k_cvt_interleave<<<dim3((unsigned)(NP * C_IN_ / 2048)), 256, 0, stream>>>(x, Xp);
  k_f32_to_bf16<<<dim3((C_INNER_ * C_IN_) / 2048), 256, 0, stream>>>(t_w, Wt);
  k_f32_to_bf16<<<dim3((C_INNER_ * C_IN_) / 2048), 256, 0, stream>>>(f_w, Wf);
  k_f32_to_bf16<<<dim3((C_OUT_ * C_IN_) / 2048), 256, 0, stream>>>(g_w, Wg);
  k_f32_to_bf16<<<dim3((C_IN_ * C_OUT_) / 2048), 256, 0, stream>>>(iw, Winc);

  k_conv_tfg<<<dim3(P_PIX / 128, 40, N_B), 256, 0, stream>>>(
      Xp, Wt, Wf, Wg, t_b, f_b, g_b, T, F, G);
  k_row_scores<<<dim3(H_DIM, N_B), 256, 0, stream>>>(T, F, Erow);
  k_col_scores<<<dim3(W_DIM, N_B), 256, 0, stream>>>(T, F, Ecol);
  k_softmax<<<dim3((unsigned)(NP / 8)), 256, 0, stream>>>(Erow, Ecol);
  k_apply_row<<<dim3(H_DIM, N_B), 256, 0, stream>>>(G, Erow, Out);
  k_apply_col<<<dim3(W_DIM, N_B), 256, 0, stream>>>(G, Ecol, Out);
  k_cvt_interleave<<<dim3((unsigned)(NP * C_OUT_ / 2048)), 256, 0, stream>>>(Out, Op);
  k_final<<<dim3(P_PIX / 128, C_IN_ / 16, N_B), 256, 0, stream>>>(x, Winc, ibv, Op, y);
}
